// GraphAttentionLayer_27616639713304
// MI455X (gfx1250) — matrix operand order repaired and checked by reading, not compiled, not run on hardware
//
#include <hip/hip_runtime.h>

// GAT layer, fused flash-style:
//   K1: h = x@W.T (f32), s_src/s_dst (pre-scaled by log2e), hT in bf16 (transposed)
//   K2: per 32-row block x 1024-col chunk: P tile generated in-register in WMMA A layout,
//       C += P@H via v_wmma_f32_16x16x32_bf16; row sums via extra WMMA vs all-ones B.
//   K3: combine column-chunk partials, divide, ELU.
// HBM floor = adj (256MB) ~ 11us @ 23.3TB/s; hT (1MB) stays in 192MB L2 and is shared
// across the 4 waves of a block (same chunk) so repeats hit L0.

#define NN 8192
#define F_IN 128
#define F_OUT 64
#define CSPLIT 8
#define CCHUNK (NN / CSPLIT)   // 1024 cols per chunk
#define CT 32                  // cols per WMMA step
#define STEPS (CCHUNK / CT)    // 32
#define LOG2E 1.44269504088896340736f

typedef __attribute__((ext_vector_type(16))) __bf16          v16bf;
typedef __attribute__((ext_vector_type(8)))  unsigned int    u32x8;
typedef __attribute__((ext_vector_type(4)))  unsigned int    u32x4;
typedef __attribute__((ext_vector_type(8)))  float           v8f;
typedef __attribute__((ext_vector_type(4)))  int             i32x4;
typedef __attribute__((ext_vector_type(4)))  float           f32x4;

// Device pass: raw v_exp_f32 (exp2). Host pass (which still checks kernel bodies):
// plain exp2f so overload resolution succeeds there too.
#if defined(__HIP_DEVICE_COMPILE__) && __has_builtin(__builtin_amdgcn_exp2f)
#define EXP2F(x) __builtin_amdgcn_exp2f(x)
#else
#define EXP2F(x) exp2f(x)
#endif

// RNE f32->bf16 (only used in the tiny K1)
__device__ __forceinline__ unsigned short f2bf(float f) {
  unsigned int u = __builtin_bit_cast(unsigned int, f);
  u += 0x7FFFu + ((u >> 16) & 1u);
  return (unsigned short)(u >> 16);
}

// Pack two f32 into packed bf16 pair (truncation) with one v_perm_b32:
// dword = { hi.bytes[3:2], lo.bytes[3:2] }
__device__ __forceinline__ unsigned int pack_bf16_pair(float lo, float hi) {
  return __builtin_amdgcn_perm(__builtin_bit_cast(unsigned int, hi),
                               __builtin_bit_cast(unsigned int, lo),
                               0x07060302u);
}

// ---------------- Kernel 1: h, s_src, s_dst (x log2e), hT(bf16) ----------------
__global__ __launch_bounds__(256) void gat_h_kernel(
    const float* __restrict__ x, const float* __restrict__ W,
    const float* __restrict__ att_w,
    unsigned short* __restrict__ hT,
    float* __restrict__ s_src, float* __restrict__ s_dst)
{
  __shared__ float sh_s[4][F_OUT];
  __shared__ float sh_d[4][F_OUT];
  const int f  = threadIdx.x;                 // 0..63 feature
  const int ry = threadIdx.y;                 // 0..3 row in block
  const int i  = blockIdx.x * 4 + ry;         // node
  const f32x4* xr = (const f32x4*)(x + (size_t)i * F_IN);
  const f32x4* wr = (const f32x4*)(W + (size_t)f * F_IN);
  float acc = 0.0f;
#pragma unroll
  for (int k = 0; k < F_IN / 4; ++k) {
    f32x4 xv = xr[k], wv = wr[k];
    acc += xv[0]*wv[0] + xv[1]*wv[1] + xv[2]*wv[2] + xv[3]*wv[3];
  }
  hT[(size_t)f * NN + i] = f2bf(acc);
  sh_s[ry][f] = acc * att_w[f];
  sh_d[ry][f] = acc * att_w[F_OUT + f];
  __syncthreads();
  if (f == 0) {
    float a = 0.0f, b = 0.0f;
#pragma unroll
    for (int t = 0; t < F_OUT; ++t) { a += sh_s[ry][t]; b += sh_d[ry][t]; }
    // pre-scale by log2e: lrelu(c*x) = c*lrelu(x) for c>0, so exp(lrelu(s)) = exp2(lrelu(s'))
    s_src[i] = a * LOG2E;
    s_dst[i] = b * LOG2E;
  }
}

// ---------------- Kernel 2: fused attention GEMM ----------------
// One wave = 32 rows x one 1024-col chunk; all 4 waves of a block share the chunk
// (identical hT/s_dst addresses -> L0 hits), distinct row groups (distinct adj streams).
__global__ __launch_bounds__(128, 1) void gat_attn_kernel(
    const int* __restrict__ adj,
    const unsigned short* __restrict__ hT,
    const float* __restrict__ s_src, const float* __restrict__ s_dst,
    float* __restrict__ pacc, float* __restrict__ pl)
{
  const int wib   = threadIdx.x >> 5;            // wave in block 0..3
  const int lane  = threadIdx.x & 31;
  const int chunk = blockIdx.x & 7;              // shared by all waves of block
  const int rg    = (blockIdx.x >> 3) * 4 + wib; // 0..255
  const int r0    = rg * 32;
  const int cbase = chunk * CCHUNK;
  const int m  = lane & 15;                // A-row / B-col / C-col within tile
  const int kh = lane >> 4;                // K-half selector (ISA fragment layout)

  const float ss0 = s_src[r0 + m];
  const float ss1 = s_src[r0 + 16 + m];

  const v8f vz = {0.f,0.f,0.f,0.f,0.f,0.f,0.f,0.f};
  v8f c[2][4]; v8f ls[2];
#pragma unroll
  for (int rb = 0; rb < 2; ++rb) {
    ls[rb] = vz;
#pragma unroll
    for (int t = 0; t < 4; ++t) c[rb][t] = vz;
  }

  u32x8 bones;
#pragma unroll
  for (int j = 0; j < 8; ++j) bones[j] = 0x3F803F80u;   // packed bf16 {1.0,1.0}

  const size_t rowA0 = (size_t)(r0 + m) * NN;
  const size_t rowA1 = (size_t)(r0 + 16 + m) * NN;

  for (int it = 0; it < STEPS; ++it) {
    const int c0 = cbase + it * CT;

    // s_dst (already x log2e) for this lane's 16 K-cols: kh*8+{0..7}, 16+kh*8+{0..7}
    f32x4 D[4];
    D[0] = *(const f32x4*)(s_dst + c0 + kh * 8);
    D[1] = *(const f32x4*)(s_dst + c0 + kh * 8 + 4);
    D[2] = *(const f32x4*)(s_dst + c0 + 16 + kh * 8);
    D[3] = *(const f32x4*)(s_dst + c0 + 16 + kh * 8 + 4);

    i32x4 A0[4], A1[4];
    A0[0] = *(const i32x4*)(adj + rowA0 + c0 + kh * 8);
    A0[1] = *(const i32x4*)(adj + rowA0 + c0 + kh * 8 + 4);
    A0[2] = *(const i32x4*)(adj + rowA0 + c0 + 16 + kh * 8);
    A0[3] = *(const i32x4*)(adj + rowA0 + c0 + 16 + kh * 8 + 4);
    A1[0] = *(const i32x4*)(adj + rowA1 + c0 + kh * 8);
    A1[1] = *(const i32x4*)(adj + rowA1 + c0 + kh * 8 + 4);
    A1[2] = *(const i32x4*)(adj + rowA1 + c0 + 16 + kh * 8);
    A1[3] = *(const i32x4*)(adj + rowA1 + c0 + 16 + kh * 8 + 4);

    if (it + 1 < STEPS) {   // uniform branch; keeps EXEC all-ones for WMMA
      __builtin_prefetch(adj + rowA0 + c0 + CT, 0, 0);
      __builtin_prefetch(adj + rowA1 + c0 + CT, 0, 0);
    }

    // Build A fragments (P tiles) as packed bf16 dwords, ISA 16-bit 16x32 layout:
    // element e<8 -> K = 8*kh + e ; e>=8 -> K = 16 + 8*kh + (e-8)
    u32x8 ap0, ap1;
#pragma unroll
    for (int e = 0; e < 8; ++e) {
      const int q = e >> 1;
      const int j = (e & 1) * 2;
      float t0a = ss0 + D[q][j];
      float t0b = ss0 + D[q][j + 1];
      float t1a = ss1 + D[q][j];
      float t1b = ss1 + D[q][j + 1];
      t0a = fmaxf(t0a, 0.01f * t0a);   // leaky_relu (slope commutes with log2e scale)
      t0b = fmaxf(t0b, 0.01f * t0b);
      t1a = fmaxf(t1a, 0.01f * t1a);
      t1b = fmaxf(t1b, 0.01f * t1b);
      float p0a = (A0[q][j]     > 0) ? EXP2F(t0a) : 0.0f;
      float p0b = (A0[q][j + 1] > 0) ? EXP2F(t0b) : 0.0f;
      float p1a = (A1[q][j]     > 0) ? EXP2F(t1a) : 0.0f;
      float p1b = (A1[q][j + 1] > 0) ? EXP2F(t1b) : 0.0f;
      ap0[e] = pack_bf16_pair(p0a, p0b);
      ap1[e] = pack_bf16_pair(p1a, p1b);
    }

    // B fragments from transposed bf16 h: element j<8 -> K = kh*8 + j, j>=8 -> K = 16 + kh*8 + (j-8), col = t*16 + m
    u32x8 bfr[4];
#pragma unroll
    for (int t = 0; t < 4; ++t) {
      const unsigned short* hrow = hT + (size_t)(t * 16 + m) * NN + c0 + kh * 8;
      u32x4 blo = *(const u32x4*)(hrow);
      u32x4 bhi = *(const u32x4*)(hrow + 16);
      bfr[t] = __builtin_shufflevector(blo, bhi, 0, 1, 2, 3, 4, 5, 6, 7);
    }

    {
      v16bf av0 = __builtin_bit_cast(v16bf, ap0);
      v16bf av1 = __builtin_bit_cast(v16bf, ap1);
#pragma unroll
      for (int t = 0; t < 4; ++t) {
        v16bf bv = __builtin_bit_cast(v16bf, bfr[t]);
        c[0][t] = __builtin_amdgcn_wmma_f32_16x16x32_bf16(
            false, av0, false, bv, (short)0, c[0][t], false, false);
        c[1][t] = __builtin_amdgcn_wmma_f32_16x16x32_bf16(
            false, av1, false, bv, (short)0, c[1][t], false, false);
      }
      v16bf b1 = __builtin_bit_cast(v16bf, bones);
      ls[0] = __builtin_amdgcn_wmma_f32_16x16x32_bf16(
          false, av0, false, b1, (short)0, ls[0], false, false);
      ls[1] = __builtin_amdgcn_wmma_f32_16x16x32_bf16(
          false, av1, false, b1, (short)0, ls[1], false, false);
    }
  }

  // Store partials. C layout: VGPR v, lane -> row = v + 8*kh (+rb*16), col = t*16 + m.
  // ls fragment has identical layout (all N columns equal the row sum).
  const int widx = rg * CSPLIT + chunk;
  float* pa  = pacc + (size_t)widx * 32 * F_OUT;
  float* pls = pl + (size_t)widx * 32;
#pragma unroll
  for (int rb = 0; rb < 2; ++rb) {
#pragma unroll
    for (int v = 0; v < 8; ++v) {
      const int r = rb * 16 + v + 8 * kh;
#pragma unroll
      for (int t = 0; t < 4; ++t)
        pa[r * F_OUT + t * 16 + m] = c[rb][t][v];
      if (m == 0) pls[r] = ls[rb][v];
    }
  }
}

// ---------------- Kernel 3: combine chunks, normalize, ELU ----------------
__global__ __launch_bounds__(256) void gat_combine_kernel(
    const float* __restrict__ pacc, const float* __restrict__ pl,
    float* __restrict__ out)
{
  const int idx = blockIdx.x * blockDim.x + threadIdx.x;  // 0..NN*F_OUT-1
  const int i  = idx >> 6;          // node
  const int f  = idx & 63;          // feature
  const int rg = i >> 5;
  const int r  = i & 31;
  float acc = 0.0f, l = 0.0f;
#pragma unroll
  for (int s = 0; s < CSPLIT; ++s) {
    const int w = rg * CSPLIT + s;
    acc += pacc[(size_t)w * 32 * F_OUT + r * F_OUT + f];
    l   += pl[(size_t)w * 32 + r];
  }
  const float val = acc / l;                       // softmax normalization
  out[idx] = (val > 0.0f) ? val : (__expf(val) - 1.0f);   // ELU
}

// ---------------- Launch ----------------
extern "C" void kernel_launch(void* const* d_in, const int* in_sizes, int n_in,
                              void* d_out, int out_size, void* d_ws, size_t ws_size,
                              hipStream_t stream) {
  (void)in_sizes; (void)n_in; (void)out_size; (void)ws_size;
  const float* x     = (const float*)d_in[0];
  const int*   adj   = (const int*)d_in[1];
  const float* W     = (const float*)d_in[2];
  const float* att_w = (const float*)d_in[3];
  float* out = (float*)d_out;

  char* ws = (char*)d_ws;
  unsigned short* hT  = (unsigned short*)ws;                 // 64*8192*2 = 1 MB
  float* s_src = (float*)(ws + (1u << 20));                  // 32 KB
  float* s_dst = s_src + NN;                                 // 32 KB
  float* pacc  = (float*)(ws + (1u << 20) + 2 * NN * sizeof(float)); // 2048*32*64*4 = 16 MB
  float* pl    = pacc + (size_t)2048 * 32 * F_OUT;           // 2048*32*4 = 256 KB

  dim3 b1(F_OUT, 4);
  gat_h_kernel<<<NN / 4, b1, 0, stream>>>(x, W, att_w, hT, s_src, s_dst);

  // 512 blocks x 4 waves; block = 4 row groups sharing one column chunk
  gat_attn_kernel<<<512, 128, 0, stream>>>(adj, hT, s_src, s_dst, pacc, pl);

  gat_combine_kernel<<<(NN * F_OUT) / 256, 256, 0, stream>>>(pacc, pl, out);
}
